// MultiheadAttention_68702296866936
// MI455X (gfx1250) — matrix operand order repaired and checked by reading, not compiled, not run on hardware
//
#include <hip/hip_runtime.h>
#include <hip/hip_bf16.h>

typedef __bf16 bf16_t;
typedef __attribute__((ext_vector_type(16))) __bf16 v16bf;
typedef __attribute__((ext_vector_type(8)))  __bf16 v8bf;
typedef __attribute__((ext_vector_type(8)))  float  v8f;
typedef unsigned int u32x4 __attribute__((ext_vector_type(4)));
typedef int          i32x4 __attribute__((ext_vector_type(4)));
typedef int          i32x8 __attribute__((ext_vector_type(8)));

#define EMBED 768
#define NHEAD 12
#define HDIM  64
#define NTOK  1024   // H*W = 32*32
#define BATCH 8
#define BNH   96     // BATCH*NHEAD
#define SCALE 0.125f // HDIM^-0.5
#define NSTEP 32     // key steps of 32 keys each

#if __has_builtin(__builtin_amdgcn_tensor_load_to_lds) && \
    __has_builtin(__builtin_amdgcn_s_wait_tensorcnt)
#define HAVE_TDM 1
#else
#define HAVE_TDM 0
#endif

__device__ __forceinline__ v8f v8f_zero() {
  v8f z;
#pragma unroll
  for (int i = 0; i < 8; ++i) z[i] = 0.f;
  return z;
}

__device__ __forceinline__ v8f wmma_bf16(v16bf a, v16bf b, v8f c) {
  // D = A(16x32 bf16) x B(32x16 bf16) + C(16x16 f32)
  return __builtin_amdgcn_wmma_f32_16x16x32_bf16(false, a, false, b, (short)0, c,
                                                 false, false);
}

#if HAVE_TDM
// Tensor Data Mover: DMA a 2D tile (rows x rowlen bf16, row-major, packed) from
// global memory into LDS.  D# packing per CDNA5 ISA.
// Wave-scalar: one issue moves the whole tile; tracked by TENSORcnt.
__device__ __forceinline__ void tdm_load_tile_2d(const bf16_t* gsrc,
                                                 unsigned lds_addr,
                                                 unsigned rows, unsigned rowlen) {
  unsigned long long ga = (unsigned long long)(uintptr_t)gsrc;
  u32x4 g0;
  g0[0] = 1u;                                            // count=1, user mode
  g0[1] = lds_addr;                                      // LDS byte address
  g0[2] = (unsigned)(ga & 0xFFFFFFFFu);                  // global_addr lo
  g0[3] = (unsigned)((ga >> 32) & 0x01FFFFFFu) | (2u << 30);  // hi25 | type=2
  i32x8 g1;
  g1[0] = (int)(1u << 16);                               // data_size = 2 bytes
  g1[1] = (int)((rowlen & 0xFFFFu) << 16);               // tensor_dim0 lo16
  g1[2] = (int)((rowlen >> 16) | ((rows & 0xFFFFu) << 16));   // dim0 hi | dim1 lo
  g1[3] = (int)((rows >> 16) | (rowlen << 16));          // dim1 hi | tile_dim0
  g1[4] = (int)rows;                                     // tile_dim1 ; tile_dim2=0
  g1[5] = (int)rowlen;                                   // tensor_dim0_stride lo32
  g1[6] = 0;                                             // stride0 hi | stride1 lo
  g1[7] = 0;
  i32x4 z4;
#pragma unroll
  for (int i = 0; i < 4; ++i) z4[i] = 0;
#if defined(__clang_major__) && (__clang_major__ >= 23)
  i32x8 z8;
#pragma unroll
  for (int i = 0; i < 8; ++i) z8[i] = 0;
  __builtin_amdgcn_tensor_load_to_lds(g0, g1, z4, z4, z8, 0);
#else
  __builtin_amdgcn_tensor_load_to_lds(g0, g1, z4, z4, 0);
#endif
}
#endif  // HAVE_TDM

// A-fragment (16x32) from fp32 row-major matrix, converting to bf16.
// Layout: lane<16 row=lane holds K 0..15; lane>=16 holds K 16..31 (contiguous).
__device__ __forceinline__ v16bf afrag_f32(const float* __restrict__ X, int row0,
                                           int ld, int e0, int lane) {
  int m = lane & 15, hi = lane >> 4;
  const float* r = X + (size_t)(row0 + m) * ld + e0 + 16 * hi;
  v16bf a;
#pragma unroll
  for (int i = 0; i < 8; ++i) a[i] = (__bf16)r[i];
#pragma unroll
  for (int i = 0; i < 8; ++i) a[8 + i] = (__bf16)r[8 + i];
  return a;
}

// B-fragment (32x16) for y = x @ W^T: B(e,f) = W[f][e]; lane n=lane&15 holds
// 16 contiguous e-values starting at 16*(lane>>4) -> contiguous row chunk of W.
__device__ __forceinline__ v16bf bfrag_wT(const float* __restrict__ W, int f0,
                                          int ld, int e0, int lane) {
  int n = lane & 15, k0 = 16 * (lane >> 4);
  const float* r = W + (size_t)(f0 + n) * ld + e0 + k0;
  v16bf b;
#pragma unroll
  for (int i = 0; i < 16; ++i) b[i] = (__bf16)r[i];
  return b;
}

// ---------------------------------------------------------------------------
// Kernel 1: fused QKV projection.  grid=(512 token-tiles, 12 f-chunks, 3), block=32
// ---------------------------------------------------------------------------
__global__ __launch_bounds__(32) void qkv_proj_kernel(
    const float* __restrict__ x, const float* __restrict__ Wq,
    const float* __restrict__ bq, const float* __restrict__ Wk,
    const float* __restrict__ Wv, const float* __restrict__ bv,
    bf16_t* __restrict__ qb, bf16_t* __restrict__ kb, bf16_t* __restrict__ vb) {
  int lane = threadIdx.x;
  int t0 = blockIdx.x * 16;
  int f0 = blockIdx.y * 64;
  int which = blockIdx.z;
  const float* W = (which == 0) ? Wq : (which == 1) ? Wk : Wv;
  const float* bias = (which == 0) ? bq : (which == 2) ? bv : nullptr;
  bf16_t* out = (which == 0) ? qb : (which == 1) ? kb : vb;

  v8f acc[4];
#pragma unroll
  for (int i = 0; i < 4; ++i) acc[i] = v8f_zero();

  for (int e0 = 0; e0 < EMBED; e0 += 32) {
    v16bf a = afrag_f32(x, t0, EMBED, e0, lane);
#pragma unroll
    for (int fc = 0; fc < 4; ++fc) {
      v16bf b = bfrag_wT(W, f0 + fc * 16, EMBED, e0, lane);
      acc[fc] = wmma_bf16(a, b, acc[fc]);
    }
  }
  int hi = lane >> 4, nl = lane & 15;
#pragma unroll
  for (int fc = 0; fc < 4; ++fc) {
    int f = f0 + fc * 16 + nl;
    float bb = bias ? bias[f] : 0.f;
    int head = f >> 6, d = f & 63;
#pragma unroll
    for (int j = 0; j < 8; ++j) {
      int t = t0 + j + 8 * hi;               // global token
      int batch = t >> 10, n = t & (NTOK - 1);
      out[(((size_t)(batch * NHEAD + head)) * NTOK + n) * HDIM + d] =
          (__bf16)(acc[fc][j] + bb);
    }
  }
}

// ---------------------------------------------------------------------------
// Kernel 2: decomposed rel-pos tables.  grid=(96*1024*32/256, 2), block=256
// ---------------------------------------------------------------------------
__global__ __launch_bounds__(256) void relpos_kernel(
    const bf16_t* __restrict__ qb, const float* __restrict__ rel_pos_h,
    const float* __restrict__ rel_pos_w, float* __restrict__ relh,
    float* __restrict__ relw) {
  int idx = blockIdx.x * 256 + threadIdx.x;
  int isW = blockIdx.y;
  int kk = idx & 31;
  int n = (idx >> 5) & (NTOK - 1);
  int bh = idx >> 15;
  if (bh >= BNH) return;
  int pos = isW ? (n & 31) : (n >> 5);
  const float* tab = (isW ? rel_pos_w : rel_pos_h) + (size_t)(pos - kk + 31) * HDIM;
  const bf16_t* qr = qb + ((size_t)bh * NTOK + n) * HDIM;
  float s = 0.f;
#pragma unroll
  for (int c = 0; c < HDIM; ++c) s += (float)qr[c] * tab[c];
  (isW ? relw : relh)[idx] = s;
}

// ---------------------------------------------------------------------------
// Kernel 3: flash attention.  grid=(96 bh, 16 qblocks), block=128 (4 waves).
// Double-buffered TDM staging: wave 0 waits TENSORcnt for the current tile,
// then issues the next tile's DMA before compute -> DMA overlaps the 8-WMMA
// + softmax body.  attn_bias is a read-once 402MB stream -> non-temporal.
// ---------------------------------------------------------------------------
__global__ __launch_bounds__(128) void flash_attn_kernel(
    const bf16_t* __restrict__ qb, const bf16_t* __restrict__ kb,
    const bf16_t* __restrict__ vb, const float* __restrict__ attn_bias,
    const float* __restrict__ relh, const float* __restrict__ relw,
    float* __restrict__ xattn) {
  __shared__ bf16_t kbuf[2][32 * 64];   // [buf][key][channel] row-major
  __shared__ bf16_t vbuf[2][32 * 64];   // [buf][key][channel] row-major
  __shared__ bf16_t pbuf[4][16 * 32];   // per-wave P tile bounce buffer

  int tid = threadIdx.x;
  int lane = tid & 31, wid = tid >> 5;
  int bh = blockIdx.x;
  int qt = blockIdx.y * 4 + wid;        // 16-query tile index (0..63)
  int batch = bh / NHEAD, head = bh % NHEAD;
  int hi = lane >> 4, nl = lane & 15;

  // Q A-fragments held in registers for the whole K loop (16x64 = two 16x32).
  v16bf aq[2];
  {
    const bf16_t* r = qb + ((size_t)bh * NTOK + qt * 16 + nl) * HDIM;
    int k0a = 16 * hi;
#pragma unroll
    for (int f = 0; f < 2; ++f) {
      v8bf lo = *(const v8bf*)(r + f * 32 + k0a);
      v8bf hi8 = *(const v8bf*)(r + f * 32 + 8 + k0a);
      v16bf a;
#pragma unroll
      for (int i = 0; i < 8; ++i) { a[i] = lo[i]; a[8 + i] = hi8[i]; }
      aq[f] = a;
    }
  }

  v8f o[4];
#pragma unroll
  for (int i = 0; i < 4; ++i) o[i] = v8f_zero();
  float mrun[8], lrun[8];
#pragma unroll
  for (int j = 0; j < 8; ++j) { mrun[j] = -1e30f; lrun[j] = 0.f; }

  const bf16_t* kbase = kb + (size_t)bh * NTOK * HDIM;
  const bf16_t* vbase = vb + (size_t)bh * NTOK * HDIM;
  const float* biasbase = attn_bias + (size_t)bh * NTOK * NTOK;
  const float* rhbase = relh + (size_t)bh * NTOK * 32;
  const float* rwbase = relw + (size_t)bh * NTOK * 32;

  // rel_w depends only on kw = g*16 + nl, which is step-invariant: hoist.
  float rw0[8], rw1[8];
#pragma unroll
  for (int j = 0; j < 8; ++j) {
    int nrow = qt * 16 + j + 8 * hi;
    rw0[j] = rwbase[(size_t)nrow * 32 + nl];
    rw1[j] = rwbase[(size_t)nrow * 32 + 16 + nl];
  }

#if HAVE_TDM
  unsigned klds[2] = {(unsigned)(uintptr_t)&kbuf[0][0],
                      (unsigned)(uintptr_t)&kbuf[1][0]};
  unsigned vlds[2] = {(unsigned)(uintptr_t)&vbuf[0][0],
                      (unsigned)(uintptr_t)&vbuf[1][0]};
  if (wid == 0) {                       // prologue: stage step 0 into buffer 0
    tdm_load_tile_2d(kbase, klds[0], 32, HDIM);
    tdm_load_tile_2d(vbase, vlds[0], 32, HDIM);
  }
#endif

  for (int step = 0; step < NSTEP; ++step) {  // 32 keys per step, kh == step
    int cur = step & 1;
#if HAVE_TDM
    if (wid == 0) __builtin_amdgcn_s_wait_tensorcnt(0);
    __syncthreads();                    // tile visible; prev readers drained
    if (wid == 0 && step + 1 < NSTEP) { // overlap next DMA with compute
      tdm_load_tile_2d(kbase + (size_t)(step + 1) * 32 * HDIM, klds[cur ^ 1],
                       32, HDIM);
      tdm_load_tile_2d(vbase + (size_t)(step + 1) * 32 * HDIM, vlds[cur ^ 1],
                       32, HDIM);
    }
#else
    {  // cooperative fallback staging (row-major, both tiles)
      int key = tid >> 2;               // 0..31
      int cp = (tid & 3) * 16;          // channel part
      const bf16_t* ksrc = kbase + (size_t)(step * 32 + key) * HDIM + cp;
      *(v8bf*)(&kbuf[cur][key * 64 + cp]) = *(const v8bf*)(ksrc);
      *(v8bf*)(&kbuf[cur][key * 64 + cp + 8]) = *(const v8bf*)(ksrc + 8);
      const bf16_t* vsrc = vbase + (size_t)(step * 32 + key) * HDIM + cp;
      *(v8bf*)(&vbuf[cur][key * 64 + cp]) = *(const v8bf*)(vsrc);
      *(v8bf*)(&vbuf[cur][key * 64 + cp + 8]) = *(const v8bf*)(vsrc + 8);
    }
    __syncthreads();
#endif
    const bf16_t* kt = &kbuf[cur][0];
    const bf16_t* vt = &vbuf[cur][0];

    // S(16x32) = Q(16x64) @ K^T(64x32): two 16x16 tiles, 2 WMMAs each (K=64)
    v8f s[2];
#pragma unroll
    for (int g = 0; g < 2; ++g) {
      v16bf b0 = *(const v16bf*)(&kt[(g * 16 + nl) * 64 + 16 * hi]);
      v16bf b1 = *(const v16bf*)(&kt[(g * 16 + nl) * 64 + 32 + 16 * hi]);
      v8f acc = v8f_zero();
      acc = wmma_bf16(aq[0], b0, acc);
      acc = wmma_bf16(aq[1], b1, acc);
      s[g] = acc;
    }

    // scale + bias (non-temporal stream) + rel-pos, online softmax update.
    // C layout: VGPR j holds row j (lanes 0-15) / row j+8 (lanes 16-31).
#pragma unroll
    for (int j = 0; j < 8; ++j) {
      int nrow = qt * 16 + j + 8 * hi;
      size_t brow = (size_t)nrow * NTOK + step * 32;
      float rh = rhbase[(size_t)nrow * 32 + step];
      float f0 = s[0][j] * SCALE + __builtin_nontemporal_load(&biasbase[brow + nl]) +
                 rh + rw0[j];
      float f1 = s[1][j] * SCALE +
                 __builtin_nontemporal_load(&biasbase[brow + 16 + nl]) + rh + rw1[j];
      float loc = fmaxf(f0, f1);
#pragma unroll
      for (int m = 1; m < 16; m <<= 1) loc = fmaxf(loc, __shfl_xor(loc, m, 16));
      float newm = fmaxf(mrun[j], loc);
      float alpha = __expf(mrun[j] - newm);
      mrun[j] = newm;
      float e0 = __expf(f0 - newm), e1 = __expf(f1 - newm);
      float rs = e0 + e1;
#pragma unroll
      for (int m = 1; m < 16; m <<= 1) rs += __shfl_xor(rs, m, 16);
      lrun[j] = lrun[j] * alpha + rs;
#pragma unroll
      for (int cc = 0; cc < 4; ++cc) o[cc][j] *= alpha;
      pbuf[wid][(j + 8 * hi) * 32 + nl] = (__bf16)e0;
      pbuf[wid][(j + 8 * hi) * 32 + 16 + nl] = (__bf16)e1;
    }

    // O(16x64) += P(16x32) @ V(32x64): P via LDS into A layout, 4 WMMAs.
    v16bf ap;
    {
      int k0a = 16 * hi;
      const bf16_t* pr = &pbuf[wid][nl * 32];
      v8bf lo = *(const v8bf*)(pr + k0a);
      v8bf hi8 = *(const v8bf*)(pr + 8 + k0a);
#pragma unroll
      for (int i = 0; i < 8; ++i) { ap[i] = lo[i]; ap[8 + i] = hi8[i]; }
    }
    int k0 = 16 * hi;  // B-fragment: lane holds 16 consecutive keys, fixed chan
#pragma unroll
    for (int cc = 0; cc < 4; ++cc) {
      int c = cc * 16 + nl;
      v16bf bvf;
#pragma unroll
      for (int i = 0; i < 16; ++i) bvf[i] = vt[(k0 + i) * 64 + c];
      o[cc] = wmma_bf16(ap, bvf, o[cc]);
    }
    __syncthreads();                    // readers drained before buf reuse
  }

  // epilogue: 1/l normalize, scatter to (b, n, head*64+d) fp32 (write-once -> NT)
#pragma unroll
  for (int j = 0; j < 8; ++j) {
    float inv = 1.0f / lrun[j];
    int nrow = qt * 16 + j + 8 * hi;
    size_t trow = ((size_t)batch * NTOK + nrow) * EMBED + head * HDIM;
#pragma unroll
    for (int cc = 0; cc < 4; ++cc)
      __builtin_nontemporal_store(o[cc][j] * inv, &xattn[trow + cc * 16 + nl]);
  }
}

// ---------------------------------------------------------------------------
// Kernel 4: LayerNorm over E=768, one wave per token, in place.
// ---------------------------------------------------------------------------
__global__ __launch_bounds__(256) void layernorm_kernel(
    float* __restrict__ x, const float* __restrict__ g,
    const float* __restrict__ b) {
  int wid = threadIdx.x >> 5, lane = threadIdx.x & 31;
  int tok = blockIdx.x * 8 + wid;
  float* row = x + (size_t)tok * EMBED;
  float vals[24];
  float s = 0.f;
#pragma unroll
  for (int i = 0; i < 24; ++i) { vals[i] = row[lane + i * 32]; s += vals[i]; }
#pragma unroll
  for (int m = 1; m < 32; m <<= 1) s += __shfl_xor(s, m, 32);
  float mean = s * (1.0f / EMBED);
  float vs = 0.f;
#pragma unroll
  for (int i = 0; i < 24; ++i) { float d = vals[i] - mean; vs += d * d; }
#pragma unroll
  for (int m = 1; m < 32; m <<= 1) vs += __shfl_xor(vs, m, 32);
  float rstd = rsqrtf(vs * (1.0f / EMBED) + 1e-5f);
#pragma unroll
  for (int i = 0; i < 24; ++i) {
    int c = lane + i * 32;
    row[c] = (vals[i] - mean) * rstd * g[c] + b[c];
  }
}

// ---------------------------------------------------------------------------
// Kernel 5: output projection y = x @ Wo^T + bo, fp32 out.  grid=(512,12), block=32
// ---------------------------------------------------------------------------
__global__ __launch_bounds__(32) void out_proj_kernel(
    const float* __restrict__ x, const float* __restrict__ Wo,
    const float* __restrict__ bo, float* __restrict__ out) {
  int lane = threadIdx.x;
  int t0 = blockIdx.x * 16;
  int f0 = blockIdx.y * 64;
  v8f acc[4];
#pragma unroll
  for (int i = 0; i < 4; ++i) acc[i] = v8f_zero();
  for (int e0 = 0; e0 < EMBED; e0 += 32) {
    v16bf a = afrag_f32(x, t0, EMBED, e0, lane);
#pragma unroll
    for (int fc = 0; fc < 4; ++fc) {
      v16bf bfr = bfrag_wT(Wo, f0 + fc * 16, EMBED, e0, lane);
      acc[fc] = wmma_bf16(a, bfr, acc[fc]);
    }
  }
  int hi = lane >> 4, nl = lane & 15;
#pragma unroll
  for (int fc = 0; fc < 4; ++fc) {
    int f = f0 + fc * 16 + nl;
    float bb = bo[f];
#pragma unroll
    for (int j = 0; j < 8; ++j) {
      int t = t0 + j + 8 * hi;
      __builtin_nontemporal_store(acc[fc][j] + bb, &out[(size_t)t * EMBED + f]);
    }
  }
}

// ---------------------------------------------------------------------------
extern "C" void kernel_launch(void* const* d_in, const int* in_sizes, int n_in,
                              void* d_out, int out_size, void* d_ws,
                              size_t ws_size, hipStream_t stream) {
  (void)in_sizes; (void)n_in; (void)out_size; (void)ws_size;
  const float* query     = (const float*)d_in[0];
  const float* attn_bias = (const float*)d_in[1];
  const float* Wq        = (const float*)d_in[2];
  const float* bq        = (const float*)d_in[3];
  const float* Wk        = (const float*)d_in[4];
  const float* Wv        = (const float*)d_in[5];
  const float* bv        = (const float*)d_in[6];
  const float* Wo        = (const float*)d_in[7];
  const float* bo        = (const float*)d_in[8];
  const float* ln_g      = (const float*)d_in[9];
  const float* ln_b      = (const float*)d_in[10];
  const float* rel_pos_h = (const float*)d_in[11];
  const float* rel_pos_w = (const float*)d_in[12];
  float* out = (float*)d_out;

  char* ws = (char*)d_ws;
  size_t off = 0;
  const size_t qkv_bytes = (size_t)BNH * NTOK * HDIM * sizeof(bf16_t);
  const size_t rel_bytes = (size_t)BNH * NTOK * 32 * sizeof(float);
  bf16_t* qb = (bf16_t*)(ws + off); off += qkv_bytes;
  bf16_t* kb = (bf16_t*)(ws + off); off += qkv_bytes;
  bf16_t* vb = (bf16_t*)(ws + off); off += qkv_bytes;
  float* relh = (float*)(ws + off); off += rel_bytes;
  float* relw = (float*)(ws + off); off += rel_bytes;
  float* xattn = (float*)(ws + off); off += (size_t)BATCH * NTOK * EMBED * sizeof(float);

  qkv_proj_kernel<<<dim3(512, 12, 3), 32, 0, stream>>>(query, Wq, bq, Wk, Wv,
                                                       bv, qb, kb, vb);
  relpos_kernel<<<dim3((BNH * NTOK * 32) / 256, 2), 256, 0, stream>>>(
      qb, rel_pos_h, rel_pos_w, relh, relw);
  flash_attn_kernel<<<dim3(BNH, 16), 128, 0, stream>>>(qb, kb, vb, attn_bias,
                                                       relh, relw, xattn);
  layernorm_kernel<<<1024, 256, 0, stream>>>(xattn, ln_g, ln_b);
  out_proj_kernel<<<dim3(512, 12), 32, 0, stream>>>(xattn, Wo, bo, out);
}
